// Yolact_16741782520355
// MI455X (gfx1250) — compile-verified, matrix operand order checked by reading, not run on hardware
//
#include <hip/hip_runtime.h>
#include <hip/hip_bf16.h>

typedef _Float16 half_t;
typedef __attribute__((ext_vector_type(16))) _Float16 v16h;
typedef __attribute__((ext_vector_type(8)))  float    v8f;
typedef __attribute__((ext_vector_type(4)))  _Float16 h4;

#define ACT_NONE 0
#define ACT_RELU 1
#define ACT_TANH 2

#define NB 8
#define NP 19248
#define NCLS 80
#define TOPK 200

// ---------------------------------------------------------------------------
// Layout conversion: NCHW f32 -> NHWC f16
// ---------------------------------------------------------------------------
__global__ __launch_bounds__(256) void cvt_act(const float* __restrict__ in,
                                               half_t* __restrict__ out,
                                               int N, int C, int H, int W) {
  size_t i = (size_t)blockIdx.x * blockDim.x + threadIdx.x;
  size_t tot = (size_t)N * C * H * W;
  if (i >= tot) return;
  int w = (int)(i % W); size_t t = i / W;
  int h = (int)(t % H); t /= H;
  int c = (int)(t % C); int n = (int)(t / C);
  out[(((size_t)n * H + h) * W + w) * C + c] = (half_t)in[i];
}

// Weights OIHW f32 -> K-chunked f16: out[((K/32)*Cout + o)*32 + K%32],
// K = (ky*KW + kx)*Cin + ci  (matches NHWC im2col K order, Cin % 32 == 0)
__global__ __launch_bounds__(256) void cvt_w(const float* __restrict__ in,
                                             half_t* __restrict__ out,
                                             int O, int Ci, int KH, int KW) {
  size_t i = (size_t)blockIdx.x * blockDim.x + threadIdx.x;
  size_t tot = (size_t)O * Ci * KH * KW;
  if (i >= tot) return;
  int kx = (int)(i % KW); size_t t = i / KW;
  int ky = (int)(t % KH); t /= KH;
  int ci = (int)(t % Ci); int o = (int)(t / Ci);
  int K = (ky * KW + kx) * Ci + ci;
  int kc = K >> 5, kk = K & 31;
  out[((size_t)kc * O + o) * 32 + kk] = (half_t)in[i];
}

// ---------------------------------------------------------------------------
// Implicit-GEMM conv with V_WMMA_F32_16X16X32_F16.
// M = N*Ho*Wo (output positions), N = Cout, K = Cin*KH*KW (Cin % 32 == 0).
// Each wave computes a 32x64 output tile (2 M-tiles x 4 N-tiles):
//   - each A fragment feeds 4 WMMAs, each B fragment feeds 2 WMMAs
//   - 24 b64 loads per K-chunk drive 8 WMMAs (3 loads/WMMA)
// K-loop is ky/kx/cslice nested -> no division in the hot loop; A/B
// addresses advance by pointer increments.
// ---------------------------------------------------------------------------
__global__ __launch_bounds__(256) void conv_wmma(
    const half_t* __restrict__ x, const half_t* __restrict__ wp,
    const float* __restrict__ bias, half_t* __restrict__ y,
    int Nb, int H, int W, int Cin, int Ho, int Wo, int Cout,
    int KH, int KW, int stride, int pad,
    int yStrideN, int yOff, int act, int tiles, int Ntiles4) {
  int wave = blockIdx.x * (blockDim.x >> 5) + ((int)threadIdx.x >> 5);
  if (wave >= tiles) return;
  int lane = (int)threadIdx.x & 31;
  int tm = wave / Ntiles4;
  int tn = wave - tm * Ntiles4;
  int HWo = Ho * Wo;
  int Mtot = Nb * HWo;
  int l15 = lane & 15;

  // Two A-row groups: rows tm*32 + l15 and +16
  int  Mr[2];
  bool mv[2];
  int  nimg[2], oy[2], ox[2];
  Mr[0] = tm * 32 + l15;
  Mr[1] = Mr[0] + 16;
#pragma unroll
  for (int u = 0; u < 2; ++u) {
    mv[u] = (Mr[u] < Mtot);
    nimg[u] = 0; oy[u] = 0; ox[u] = 0;
    if (mv[u]) {
      nimg[u] = Mr[u] / HWo;
      int r = Mr[u] - nimg[u] * HWo;
      oy[u] = r / Wo;
      ox[u] = r - oy[u] * Wo;
    }
  }

  int nc0 = tn * 64 + l15;           // first of 4 column blocks (+0,+16,+32,+48)
  bool nvld[4];
#pragma unroll
  for (int j = 0; j < 4; ++j) nvld[j] = (nc0 + 16 * j) < Cout;

  int kb = (lane < 16) ? 0 : 8;      // K sub-block per lane half
  const int KO[4] = {kb, kb + 4, kb + 16, kb + 20};

  v8f acc[2][4];
#pragma unroll
  for (int u = 0; u < 2; ++u)
#pragma unroll
    for (int j = 0; j < 4; ++j) acc[u][j] = (v8f){};

  const h4 zero4 = {};
  int cchunks = Cin >> 5;                 // 32-channel slices per tap
  size_t bstep = (size_t)Cout * 32;       // halves per K-chunk row of packed W

  for (int ky = 0; ky < KH; ++ky) {
    for (int kx = 0; kx < KW; ++kx) {
      bool inb[2];
      const half_t* ap[2];
#pragma unroll
      for (int u = 0; u < 2; ++u) {
        int iy = oy[u] * stride - pad + ky;
        int ix = ox[u] * stride - pad + kx;
        inb[u] = mv[u] && (iy >= 0) && (iy < H) && (ix >= 0) && (ix < W);
        int cy = inb[u] ? iy : 0, cx = inb[u] ? ix : 0;
        ap[u] = x + (((size_t)nimg[u] * H + cy) * W + cx) * (size_t)Cin;
      }
      const half_t* bp =
          wp + (size_t)((ky * KW + kx) * cchunks) * bstep + (size_t)nc0 * 32;
      for (int cs = 0; cs < cchunks; ++cs) {
        v16h a[2], b[4];
#pragma unroll
        for (int g = 0; g < 4; ++g) {
          int ko = KO[g];
#pragma unroll
          for (int u = 0; u < 2; ++u) {
            h4 av = inb[u] ? *(const h4*)(ap[u] + ko) : zero4;
            a[u][4 * g + 0] = av[0]; a[u][4 * g + 1] = av[1];
            a[u][4 * g + 2] = av[2]; a[u][4 * g + 3] = av[3];
          }
#pragma unroll
          for (int j = 0; j < 4; ++j) {
            h4 bv = nvld[j] ? *(const h4*)(bp + (size_t)(16 * j) * 32 + ko)
                            : zero4;
            b[j][4 * g + 0] = bv[0]; b[j][4 * g + 1] = bv[1];
            b[j][4 * g + 2] = bv[2]; b[j][4 * g + 3] = bv[3];
          }
        }
#pragma unroll
        for (int u = 0; u < 2; ++u)
#pragma unroll
          for (int j = 0; j < 4; ++j)
            acc[u][j] = __builtin_amdgcn_wmma_f32_16x16x32_f16(
                false, a[u], false, b[j], (short)0, acc[u][j], false, false);
        ap[0] += 32;
        ap[1] += 32;
        bp += bstep;
      }
    }
  }

  // Store 32x64 tile (C/D layout: lanes 0-15 rows r, lanes 16-31 rows 8+r)
#pragma unroll
  for (int u = 0; u < 2; ++u) {
#pragma unroll
    for (int r = 0; r < 8; ++r) {
      int mloc = (lane < 16) ? r : (8 + r);
      int Ms = tm * 32 + u * 16 + mloc;
      if (Ms < Mtot) {
        int ni = Ms / HWo;
        int rr = Ms - ni * HWo;
        size_t rowbase = (size_t)ni * (size_t)yStrideN + (size_t)yOff +
                         (size_t)rr * (size_t)Cout;
#pragma unroll
        for (int j = 0; j < 4; ++j) {
          int nc = nc0 + 16 * j;
          if (nc < Cout) {
            float v = acc[u][j][r] + bias[nc];
            if (act == ACT_RELU) v = fmaxf(v, 0.f);
            else if (act == ACT_TANH) v = tanhf(v);
            y[rowbase + nc] = (half_t)v;
          }
        }
      }
    }
  }
}

// ---------------------------------------------------------------------------
// FPN top-down: dst[b,y,x,c] += bilinear(src)[b,y,x,c]  (half-pixel centers)
// ---------------------------------------------------------------------------
__global__ __launch_bounds__(256) void upsample_add(
    const half_t* __restrict__ src, half_t* __restrict__ dst,
    int Nb, int Hs, int Ws, int Hd, int Wd, int C) {
  size_t i = (size_t)blockIdx.x * blockDim.x + threadIdx.x;
  size_t tot = (size_t)Nb * Hd * Wd * C;
  if (i >= tot) return;
  int c = (int)(i % C); size_t t = i / C;
  int x = (int)(t % Wd); t /= Wd;
  int y = (int)(t % Hd); int n = (int)(t / Hd);
  float sy = (y + 0.5f) * (float)Hs / (float)Hd - 0.5f;
  float sx = (x + 0.5f) * (float)Ws / (float)Wd - 0.5f;
  int y0 = (int)floorf(sy), x0 = (int)floorf(sx);
  float fy = sy - (float)y0, fx = sx - (float)x0;
  int y1 = y0 + 1, x1 = x0 + 1;
  y0 = y0 < 0 ? 0 : (y0 > Hs - 1 ? Hs - 1 : y0);
  y1 = y1 < 0 ? 0 : (y1 > Hs - 1 ? Hs - 1 : y1);
  x0 = x0 < 0 ? 0 : (x0 > Ws - 1 ? Ws - 1 : x0);
  x1 = x1 < 0 ? 0 : (x1 > Ws - 1 ? Ws - 1 : x1);
  const size_t nb = (size_t)n * Hs;
  float v00 = (float)src[((nb + y0) * Ws + x0) * C + c];
  float v01 = (float)src[((nb + y0) * Ws + x1) * C + c];
  float v10 = (float)src[((nb + y1) * Ws + x0) * C + c];
  float v11 = (float)src[((nb + y1) * Ws + x1) * C + c];
  float v = v00 * (1.f - fy) * (1.f - fx) + v01 * (1.f - fy) * fx +
            v10 * fy * (1.f - fx) + v11 * fy * fx;
  dst[i] = (half_t)((float)dst[i] + v);
}

// ---------------------------------------------------------------------------
// Priors, decode, softmax scores
// ---------------------------------------------------------------------------
__global__ __launch_bounds__(256) void build_priors(float* __restrict__ pr) {
  int p = blockIdx.x * blockDim.x + threadIdx.x;
  if (p >= NP) return;
  const int   HS[5]  = {69, 35, 18, 9, 5};
  const float SC[5]  = {24.f, 48.f, 96.f, 192.f, 384.f};
  const int   OFF[6] = {0, 14283, 17958, 18930, 19173, NP};
  int l = 0;
  while (l < 4 && p >= OFF[l + 1]) ++l;
  int i = p - OFF[l];
  int a = i % 3;
  int cell = i / 3;
  int Wl = HS[l];
  int x = cell % Wl, y = cell / Wl;
  const float AR[3] = {1.f, 0.5f, 2.f};
  float s = SC[l];
  float cx = (x + 0.5f) / (float)Wl;
  float cy = (y + 0.5f) / (float)Wl;
  float sq = sqrtf(AR[a]);
  pr[p * 4 + 0] = cx;
  pr[p * 4 + 1] = cy;
  pr[p * 4 + 2] = s * sq / 550.f;
  pr[p * 4 + 3] = s / sq / 550.f;
}

__global__ __launch_bounds__(256) void decode_boxes(
    const half_t* __restrict__ loc, const float* __restrict__ pr,
    float* __restrict__ boxes, int Bn, int P) {
  int i = blockIdx.x * blockDim.x + threadIdx.x;
  if (i >= Bn * P) return;
  int p = i % P;
  const half_t* l = loc + (size_t)i * 4;
  float px = pr[p * 4 + 0], py = pr[p * 4 + 1];
  float pw = pr[p * 4 + 2], ph = pr[p * 4 + 3];
  float cx = px + (float)l[0] * 0.1f * pw;
  float cy = py + (float)l[1] * 0.1f * ph;
  float w = pw * expf((float)l[2] * 0.2f);
  float h = ph * expf((float)l[3] * 0.2f);
  boxes[(size_t)i * 4 + 0] = cx - w * 0.5f;
  boxes[(size_t)i * 4 + 1] = cy - h * 0.5f;
  boxes[(size_t)i * 4 + 2] = cx + w * 0.5f;
  boxes[(size_t)i * 4 + 3] = cy + h * 0.5f;
}

__global__ __launch_bounds__(256) void softmax_scores(
    const half_t* __restrict__ conf, float* __restrict__ scores,
    int Bn, int P) {
  int i = blockIdx.x * blockDim.x + threadIdx.x;
  if (i >= Bn * P) return;
  int b = i / P, p = i % P;
  const half_t* c = conf + ((size_t)b * P + p) * 81;
  float m = -1e30f;
  for (int k = 0; k < 81; ++k) m = fmaxf(m, (float)c[k]);
  float s = 0.f;
  for (int k = 0; k < 81; ++k) s += expf((float)c[k] - m);
  float inv = 1.f / s;
  for (int k = 1; k < 81; ++k)
    scores[(((size_t)b * NCLS) + (k - 1)) * P + p] = expf((float)c[k] - m) * inv;
}

// ---------------------------------------------------------------------------
// Per-(b,class) top-200 by iterative block-wide argmax on a scratch row.
// ---------------------------------------------------------------------------
__global__ __launch_bounds__(256) void topk_select(
    const float* __restrict__ scores, float* __restrict__ scratch,
    float* __restrict__ topv, int* __restrict__ topi, int P) {
  int bc = blockIdx.x;
  const float* row = scores + (size_t)bc * P;
  float* scr = scratch + (size_t)bc * P;
  for (int i = threadIdx.x; i < P; i += blockDim.x) scr[i] = row[i];
  __syncthreads();
  __shared__ float sv[256];
  __shared__ int   si[256];
  for (int k = 0; k < TOPK; ++k) {
    float best = -1e30f;
    int bi = 0;
    for (int i = threadIdx.x; i < P; i += blockDim.x) {
      float v = scr[i];
      if (v > best) { best = v; bi = i; }
    }
    sv[threadIdx.x] = best; si[threadIdx.x] = bi;
    __syncthreads();
    for (int s = 128; s > 0; s >>= 1) {
      if ((int)threadIdx.x < s && sv[threadIdx.x + s] > sv[threadIdx.x]) {
        sv[threadIdx.x] = sv[threadIdx.x + s];
        si[threadIdx.x] = si[threadIdx.x + s];
      }
      __syncthreads();
    }
    if (threadIdx.x == 0) {
      topv[(size_t)bc * TOPK + k] = sv[0];
      topi[(size_t)bc * TOPK + k] = si[0];
      scr[si[0]] = -1e30f;
    }
    __syncthreads();
  }
}

// ---------------------------------------------------------------------------
// Fast-NMS + final det write: [B,80,200,37]
// ---------------------------------------------------------------------------
__global__ __launch_bounds__(256) void nms_write(
    const float* __restrict__ boxes, const float* __restrict__ topv,
    const int* __restrict__ topi, const half_t* __restrict__ maskc,
    float* __restrict__ out, int P) {
  int bc = blockIdx.x;
  int b = bc / NCLS;
  __shared__ float sb[TOPK * 4];
  __shared__ float ss[TOPK];
  for (int i = threadIdx.x; i < TOPK; i += blockDim.x) {
    int p = topi[(size_t)bc * TOPK + i];
    ss[i] = topv[(size_t)bc * TOPK + i];
    const float* bp = boxes + ((size_t)b * P + p) * 4;
    sb[i * 4 + 0] = bp[0]; sb[i * 4 + 1] = bp[1];
    sb[i * 4 + 2] = bp[2]; sb[i * 4 + 3] = bp[3];
  }
  __syncthreads();
  int i = threadIdx.x;
  if (i < TOPK) {
    float x1 = sb[i * 4], y1 = sb[i * 4 + 1], x2 = sb[i * 4 + 2], y2 = sb[i * 4 + 3];
    float ai = fmaxf(x2 - x1, 0.f) * fmaxf(y2 - y1, 0.f);
    float mx = 0.f;
    for (int j = 0; j < i; ++j) {
      float jx1 = sb[j * 4], jy1 = sb[j * 4 + 1];
      float jx2 = sb[j * 4 + 2], jy2 = sb[j * 4 + 3];
      float aj = fmaxf(jx2 - jx1, 0.f) * fmaxf(jy2 - jy1, 0.f);
      float ix1 = fmaxf(x1, jx1), iy1 = fmaxf(y1, jy1);
      float ix2 = fminf(x2, jx2), iy2 = fminf(y2, jy2);
      float inter = fmaxf(ix2 - ix1, 0.f) * fmaxf(iy2 - iy1, 0.f);
      mx = fmaxf(mx, inter / (ai + aj - inter + 1e-9f));
    }
    float keep = (mx <= 0.5f && ss[i] > 0.05f) ? 1.f : 0.f;
    size_t base = (((size_t)bc * TOPK) + i) * 37;
    out[base + 0] = x1 * keep; out[base + 1] = y1 * keep;
    out[base + 2] = x2 * keep; out[base + 3] = y2 * keep;
    out[base + 4] = ss[i] * keep;
    int p = topi[(size_t)bc * TOPK + i];
    const half_t* mp = maskc + ((size_t)b * P + p) * 32;
    for (int k = 0; k < 32; ++k) out[base + 5 + k] = (float)mp[k] * keep;
  }
}

// ---------------------------------------------------------------------------
// Host side
// ---------------------------------------------------------------------------
static void launch_conv(hipStream_t s, const half_t* x, const half_t* wp,
                        const float* bias, half_t* y,
                        int Nb, int H, int W, int Cin, int Ho, int Wo, int Cout,
                        int KH, int KW, int stride, int pad,
                        int yStrideN, int yOff, int act) {
  int Mtot = Nb * Ho * Wo;
  int Mt = (Mtot + 31) / 32;
  int Nt4 = (Cout + 63) / 64;   // 32x64 tile per wave
  int tiles = Mt * Nt4;
  int blocks = (tiles + 7) / 8;
  conv_wmma<<<blocks, 256, 0, s>>>(x, wp, bias, y, Nb, H, W, Cin, Ho, Wo, Cout,
                                   KH, KW, stride, pad, yStrideN, yOff, act,
                                   tiles, Nt4);
}

static inline int ceil_div(size_t a, int b) { return (int)((a + b - 1) / b); }

extern "C" void kernel_launch(void* const* d_in, const int* in_sizes, int n_in,
                              void* d_out, int out_size, void* d_ws, size_t ws_size,
                              hipStream_t stream) {
  (void)in_sizes; (void)n_in; (void)out_size; (void)ws_size;

  const float* c3      = (const float*)d_in[0];
  const float* c4      = (const float*)d_in[1];
  const float* c5      = (const float*)d_in[2];
  const float* lat_w3  = (const float*)d_in[3];  const float* lat_b3 = (const float*)d_in[4];
  const float* lat_w4  = (const float*)d_in[5];  const float* lat_b4 = (const float*)d_in[6];
  const float* lat_w5  = (const float*)d_in[7];  const float* lat_b5 = (const float*)d_in[8];
  const float* pred_w3 = (const float*)d_in[9];  const float* pred_b3 = (const float*)d_in[10];
  const float* pred_w4 = (const float*)d_in[11]; const float* pred_b4 = (const float*)d_in[12];
  const float* pred_w5 = (const float*)d_in[13]; const float* pred_b5 = (const float*)d_in[14];
  const float* down_w1 = (const float*)d_in[15]; const float* down_b1 = (const float*)d_in[16];
  const float* down_w2 = (const float*)d_in[17]; const float* down_b2 = (const float*)d_in[18];
  const float* up_w    = (const float*)d_in[19]; const float* up_b   = (const float*)d_in[20];
  const float* bbox_w  = (const float*)d_in[21]; const float* bbox_b = (const float*)d_in[22];
  const float* conf_w  = (const float*)d_in[23]; const float* conf_b = (const float*)d_in[24];
  const float* mask_w  = (const float*)d_in[25]; const float* mask_b = (const float*)d_in[26];

  char* ws = (char*)d_ws;
  size_t off = 0;
  auto alloc = [&](size_t bytes) -> char* {
    char* p = ws + off;
    off += (bytes + 255) & ~(size_t)255;
    return p;
  };

  // Region A (time-aliased): [acts] -> [head temp] -> [topk scratch]
  const size_t szC3 = (size_t)2 * NB * 69 * 69 * 512;   // 39,002,112 B
  const size_t szC4 = (size_t)2 * NB * 35 * 35 * 1024;  // 20,070,400 B
  const size_t szC5 = (size_t)2 * NB * 18 * 18 * 2048;  // 10,616,832 B
  char* regionA = alloc(szC3 + szC4 + szC5);
  half_t* actC3 = (half_t*)regionA;
  half_t* actC4 = (half_t*)(regionA + szC3);
  half_t* actC5 = (half_t*)(regionA + szC3 + szC4);
  half_t* tbuf    = (half_t*)regionA;  // 19.5 MB needed; acts dead by then
  float*  scratch = (float*)regionA;   // 49.3 MB needed; tbuf dead by then

  // Packed weights (f16)
  half_t* wLat3  = (half_t*)alloc((size_t)2 * 512 * 256);
  half_t* wLat4  = (half_t*)alloc((size_t)2 * 1024 * 256);
  half_t* wLat5  = (half_t*)alloc((size_t)2 * 2048 * 256);
  half_t* wPred3 = (half_t*)alloc((size_t)2 * 2304 * 256);
  half_t* wPred4 = (half_t*)alloc((size_t)2 * 2304 * 256);
  half_t* wPred5 = (half_t*)alloc((size_t)2 * 2304 * 256);
  half_t* wDown1 = (half_t*)alloc((size_t)2 * 2304 * 256);
  half_t* wDown2 = (half_t*)alloc((size_t)2 * 2304 * 256);
  half_t* wUp    = (half_t*)alloc((size_t)2 * 2304 * 256);
  half_t* wBbox  = (half_t*)alloc((size_t)2 * 2304 * 12);
  half_t* wConf  = (half_t*)alloc((size_t)2 * 2304 * 243);
  half_t* wMask  = (half_t*)alloc((size_t)2 * 2304 * 96);

  // FPN buffers (NHWC f16, C=256)
  half_t* lat0 = (half_t*)alloc((size_t)2 * NB * 69 * 69 * 256);
  half_t* lat1 = (half_t*)alloc((size_t)2 * NB * 35 * 35 * 256);
  half_t* lat2 = (half_t*)alloc((size_t)2 * NB * 18 * 18 * 256);
  half_t* pr0  = (half_t*)alloc((size_t)2 * NB * 69 * 69 * 256);
  half_t* pr1  = (half_t*)alloc((size_t)2 * NB * 35 * 35 * 256);
  half_t* pr2  = (half_t*)alloc((size_t)2 * NB * 18 * 18 * 256);
  half_t* p6b  = (half_t*)alloc((size_t)2 * NB * 9 * 9 * 256);
  half_t* p7b  = (half_t*)alloc((size_t)2 * NB * 5 * 5 * 256);

  // Head outputs concatenated over levels: [B, P, C]
  half_t* locA  = (half_t*)alloc((size_t)2 * NB * NP * 4);
  half_t* confA = (half_t*)alloc((size_t)2 * NB * NP * 81);
  half_t* maskA = (half_t*)alloc((size_t)2 * NB * NP * 32);

  float* priors = (float*)alloc((size_t)4 * NP * 4);
  float* boxes  = (float*)alloc((size_t)4 * NB * NP * 4);
  float* scores = (float*)alloc((size_t)4 * NB * NCLS * NP);
  float* topv   = (float*)alloc((size_t)4 * NB * NCLS * TOPK);
  int*   topi   = (int*)  alloc((size_t)4 * NB * NCLS * TOPK);

  // --- Phase 1: layout conversions -----------------------------------------
  cvt_act<<<ceil_div((size_t)NB * 512 * 69 * 69, 256), 256, 0, stream>>>(c3, actC3, NB, 512, 69, 69);
  cvt_act<<<ceil_div((size_t)NB * 1024 * 35 * 35, 256), 256, 0, stream>>>(c4, actC4, NB, 1024, 35, 35);
  cvt_act<<<ceil_div((size_t)NB * 2048 * 18 * 18, 256), 256, 0, stream>>>(c5, actC5, NB, 2048, 18, 18);

  cvt_w<<<ceil_div((size_t)256 * 512, 256), 256, 0, stream>>>(lat_w3, wLat3, 256, 512, 1, 1);
  cvt_w<<<ceil_div((size_t)256 * 1024, 256), 256, 0, stream>>>(lat_w4, wLat4, 256, 1024, 1, 1);
  cvt_w<<<ceil_div((size_t)256 * 2048, 256), 256, 0, stream>>>(lat_w5, wLat5, 256, 2048, 1, 1);
  cvt_w<<<ceil_div((size_t)256 * 2304, 256), 256, 0, stream>>>(pred_w3, wPred3, 256, 256, 3, 3);
  cvt_w<<<ceil_div((size_t)256 * 2304, 256), 256, 0, stream>>>(pred_w4, wPred4, 256, 256, 3, 3);
  cvt_w<<<ceil_div((size_t)256 * 2304, 256), 256, 0, stream>>>(pred_w5, wPred5, 256, 256, 3, 3);
  cvt_w<<<ceil_div((size_t)256 * 2304, 256), 256, 0, stream>>>(down_w1, wDown1, 256, 256, 3, 3);
  cvt_w<<<ceil_div((size_t)256 * 2304, 256), 256, 0, stream>>>(down_w2, wDown2, 256, 256, 3, 3);
  cvt_w<<<ceil_div((size_t)256 * 2304, 256), 256, 0, stream>>>(up_w, wUp, 256, 256, 3, 3);
  cvt_w<<<ceil_div((size_t)12 * 2304, 256), 256, 0, stream>>>(bbox_w, wBbox, 12, 256, 3, 3);
  cvt_w<<<ceil_div((size_t)243 * 2304, 256), 256, 0, stream>>>(conf_w, wConf, 243, 256, 3, 3);
  cvt_w<<<ceil_div((size_t)96 * 2304, 256), 256, 0, stream>>>(mask_w, wMask, 96, 256, 3, 3);

  // --- Phase 2: FPN (WMMA convs) -------------------------------------------
  launch_conv(stream, actC5, wLat5, lat_b5, lat2, NB, 18, 18, 2048, 18, 18, 256, 1, 1, 1, 0, 18 * 18 * 256, 0, ACT_NONE);
  launch_conv(stream, actC4, wLat4, lat_b4, lat1, NB, 35, 35, 1024, 35, 35, 256, 1, 1, 1, 0, 35 * 35 * 256, 0, ACT_NONE);
  launch_conv(stream, actC3, wLat3, lat_b3, lat0, NB, 69, 69, 512, 69, 69, 256, 1, 1, 1, 0, 69 * 69 * 256, 0, ACT_NONE);

  upsample_add<<<ceil_div((size_t)NB * 35 * 35 * 256, 256), 256, 0, stream>>>(lat2, lat1, NB, 18, 18, 35, 35, 256);
  upsample_add<<<ceil_div((size_t)NB * 69 * 69 * 256, 256), 256, 0, stream>>>(lat1, lat0, NB, 35, 35, 69, 69, 256);

  launch_conv(stream, lat0, wPred3, pred_b3, pr0, NB, 69, 69, 256, 69, 69, 256, 3, 3, 1, 1, 69 * 69 * 256, 0, ACT_RELU);
  launch_conv(stream, lat1, wPred4, pred_b4, pr1, NB, 35, 35, 256, 35, 35, 256, 3, 3, 1, 1, 35 * 35 * 256, 0, ACT_RELU);
  launch_conv(stream, lat2, wPred5, pred_b5, pr2, NB, 18, 18, 256, 18, 18, 256, 3, 3, 1, 1, 18 * 18 * 256, 0, ACT_RELU);

  launch_conv(stream, pr2, wDown1, down_b1, p6b, NB, 18, 18, 256, 9, 9, 256, 3, 3, 2, 1, 9 * 9 * 256, 0, ACT_NONE);
  launch_conv(stream, p6b, wDown2, down_b2, p7b, NB, 9, 9, 256, 5, 5, 256, 3, 3, 2, 1, 5 * 5 * 256, 0, ACT_NONE);

  // --- Phase 3: shared prediction head over 5 levels -----------------------
  const half_t* feats[5] = {pr0, pr1, pr2, p6b, p7b};
  const int HS[5]  = {69, 35, 18, 9, 5};
  const int OFF[5] = {0, 14283, 17958, 18930, 19173};
  for (int l = 0; l < 5; ++l) {
    int Hh = HS[l];
    launch_conv(stream, feats[l], wUp, up_b, tbuf, NB, Hh, Hh, 256, Hh, Hh, 256, 3, 3, 1, 1, Hh * Hh * 256, 0, ACT_RELU);
    launch_conv(stream, tbuf, wBbox, bbox_b, locA,  NB, Hh, Hh, 256, Hh, Hh, 12,  3, 3, 1, 1, NP * 4,  OFF[l] * 4,  ACT_NONE);
    launch_conv(stream, tbuf, wConf, conf_b, confA, NB, Hh, Hh, 256, Hh, Hh, 243, 3, 3, 1, 1, NP * 81, OFF[l] * 81, ACT_NONE);
    launch_conv(stream, tbuf, wMask, mask_b, maskA, NB, Hh, Hh, 256, Hh, Hh, 96,  3, 3, 1, 1, NP * 32, OFF[l] * 32, ACT_TANH);
  }

  // --- Phase 4: decode / softmax / top-k / Fast-NMS ------------------------
  build_priors<<<ceil_div((size_t)NP, 256), 256, 0, stream>>>(priors);
  decode_boxes<<<ceil_div((size_t)NB * NP, 256), 256, 0, stream>>>(locA, priors, boxes, NB, NP);
  softmax_scores<<<ceil_div((size_t)NB * NP, 256), 256, 0, stream>>>(confA, scores, NB, NP);
  topk_select<<<NB * NCLS, 256, 0, stream>>>(scores, scratch, topv, topi, NP);
  nms_write<<<NB * NCLS, 256, 0, stream>>>(boxes, topv, topi, maskA, (float*)d_out, NP);
}
